// EdgeBlock_14035953123598
// MI455X (gfx1250) — matrix operand order repaired and checked by reading, not compiled, not run on hardware
//
#include <hip/hip_runtime.h>

typedef __attribute__((ext_vector_type(16))) _Float16 v16h;
typedef __attribute__((ext_vector_type(8)))  _Float16 half8;
typedef __attribute__((ext_vector_type(8)))  float    v8f;

#define CAT_DIM   288
#define INTER_DIM 256
#define GATE_DIMC 32

// packed-weight region offsets (in halves)
#define SIDE_STRIDE 177152
#define OFF_WCAT 0
#define OFF_INT1 65536
#define OFF_INT2 131072
#define OFF_G1   163840
#define OFF_G2   173056
#define OFF_UPD  354304
#define OFF_O1   436224
#define OFF_O2   452608

#define K2_WAVES 2
#define K3_WAVES 2

__device__ __forceinline__ v8f wmma16(v16h a, v16h b, v8f c) {
  // D = A(16x32 f16) * B(32x16 f16) + C(16x16 f32)
  return __builtin_amdgcn_wmma_f32_16x16x32_f16(false, a, false, b, (short)0, c,
                                                false, false);
}

// A fragment from an LDS row-major f16 tile (16 rows x stride cols), K-chunk kc.
// Lane l: row M = l&15; slots 0..7 -> K = kc*32 + 8g + [0..7],
//                        slots 8..15 -> K = kc*32 + 16 + 8g + [0..7]  (g = l>>4)
__device__ __forceinline__ v16h loadA_lds(const _Float16* base, int stride, int kc, int lane) {
  const int m = lane & 15, g = lane >> 4;
  const _Float16* p = base + m * stride + kc * 32 + g * 8;
  half8 lo = *(const half8*)p;
  half8 hi = *(const half8*)(p + 16);
  return __builtin_shufflevector(lo, hi, 0,1,2,3,4,5,6,7,8,9,10,11,12,13,14,15);
}

// B fragment: packed tile of 512 halves, lane reads 16 contiguous halves (32B).
__device__ __forceinline__ v16h loadB_glb(const _Float16* tile, int lane) {
  return *(const v16h*)(tile + lane * 16);
}

__global__ void zero_f32(float* p, size_t n) {
  size_t i = (size_t)blockIdx.x * blockDim.x + threadIdx.x;
  if (i < n) p[i] = 0.0f;
}

// Pack W (Ksrc x N, row-major f32) into f16 B fragments.
// Dest tile index = nb*ktotChunks + (kbOff+kb); element (lane l, slot h):
//   value = W[kb*32 + (h<8 ? 8*(l>>4)+h : 16+8*(l>>4)+(h-8))][nb*16 + (l&15)]
__global__ void pack_w(const float* __restrict__ src, _Float16* __restrict__ dst,
                       int Ksrc, int N, int ktotChunks, int kbOff) {
  int t = blockIdx.x * 256 + threadIdx.x;
  int total = Ksrc * N;
  if (t >= total) return;
  int h = t & 15;
  int l = (t >> 4) & 31;
  int tl = t >> 9;
  int kch = Ksrc >> 5;
  int kb = tl % kch;
  int nb = tl / kch;
  int srcRow = kb * 32 + (h < 8 ? 8 * (l >> 4) + h : 16 + 8 * (l >> 4) + (h - 8));
  int srcCol = nb * 16 + (l & 15);
  dst[(size_t)(nb * ktotChunks + kbOff + kb) * 512 + l * 16 + h] =
      (_Float16)src[(size_t)srcRow * N + srcCol];
}

struct P2 {
  const float *h_bond, *h_node, *extra;
  const int *left, *right;
  const _Float16 *w;
  const float *il1b[2], *ilng[2], *ilnb[2], *il2b[2];
  const float *gl1b[2], *glng[2], *glnb[2], *gl2b[2];
  float *accumL, *accumR;
  int E, numTiles;
};

__global__ void __launch_bounds__(32 * K2_WAVES) edge_msg_kernel(P2 p) {
  __shared__ _Float16 sX[K2_WAVES][16 * CAT_DIM];
  __shared__ _Float16 sH[K2_WAVES][16 * INTER_DIM];
  __shared__ _Float16 sG[K2_WAVES][16 * GATE_DIMC];
  const int lane = threadIdx.x & 31;
  const int wv = threadIdx.x >> 5;
  const int tile = blockIdx.x * K2_WAVES + wv;
  const bool valid = tile < p.numTiles;
  const int e0 = valid ? tile * 16 : 0;
  _Float16* X = sX[wv];
  _Float16* H = sH[wv];
  _Float16* G = sG[wv];
  const int m15 = lane & 15, g2 = lane >> 4;

  for (int s = 0; s < 2; ++s) {
    const int* nIdx = (s == 0) ? p.left : p.right;   // node feeding this ffn
    const int* dIdx = (s == 0) ? p.right : p.left;   // segment destination
    float* accum = (s == 0) ? p.accumL : p.accumR;
    const _Float16* W = p.w + s * SIDE_STRIDE;

    // stage concat input [h_bond | h_node[idx] | extra] as f16 (16 x 288)
    for (int i = lane; i < 16 * 128; i += 32) {
      int r = i >> 7, c = i & 127;
      int e = min(e0 + r, p.E - 1);
      X[r * CAT_DIM + c] = (_Float16)p.h_bond[(size_t)e * 128 + c];
      int nid = nIdx[e];
      X[r * CAT_DIM + 128 + c] = (_Float16)p.h_node[(size_t)nid * 128 + c];
    }
    for (int i = lane; i < 16 * 32; i += 32) {
      int r = i >> 5, c = i & 31;
      int e = min(e0 + r, p.E - 1);
      X[r * CAT_DIM + 256 + c] = (_Float16)p.extra[(size_t)e * 32 + c];
    }
    __syncthreads();

    // ---- gate l1: (16x288)@(288x32) + b ----
    #pragma unroll
    for (int nt = 0; nt < 2; ++nt) {
      v8f acc = {};
      #pragma unroll
      for (int kc = 0; kc < 9; ++kc) {
        v16h a = loadA_lds(X, CAT_DIM, kc, lane);
        v16h b = loadB_glb(W + OFF_G1 + (nt * 9 + kc) * 512, lane);
        acc = wmma16(a, b, acc);
      }
      int col = nt * 16 + m15;
      float bv = p.gl1b[s][col];
      #pragma unroll
      for (int r = 0; r < 8; ++r)
        G[(r + 8 * g2) * GATE_DIMC + col] = (_Float16)(acc[r] + bv);
    }
    __syncthreads();
    // gate LN + relu (rows in lanes 0..15)
    if (lane < 16) {
      float s1 = 0.f, s2 = 0.f;
      for (int c = 0; c < 32; ++c) { float v = (float)G[lane * 32 + c]; s1 += v; s2 += v * v; }
      float mu = s1 * (1.f / 32.f), var = s2 * (1.f / 32.f) - mu * mu;
      float rs = rsqrtf(var + 1e-5f);
      for (int c = 0; c < 32; ++c) {
        float v = ((float)G[lane * 32 + c] - mu) * rs * p.glng[s][c] + p.glnb[s][c];
        G[lane * 32 + c] = (_Float16)(v > 0.f ? v : 0.f);
      }
    }
    __syncthreads();
    // ---- gate l2 (K=32) + sigmoid -> registers ----
    v8f gsig[8];
    {
      v16h ag = loadA_lds(G, GATE_DIMC, 0, lane);
      #pragma unroll
      for (int nt = 0; nt < 8; ++nt) {
        v8f acc = {};
        v16h b = loadB_glb(W + OFF_G2 + nt * 512, lane);
        acc = wmma16(ag, b, acc);
        float bv = p.gl2b[s][nt * 16 + m15];
        #pragma unroll
        for (int r = 0; r < 8; ++r) {
          float x = acc[r] + bv;
          gsig[nt][r] = 1.f / (1.f + __expf(-x));
        }
      }
    }

    // ---- inter0 = [h_bond|hn] @ Wcat (256->256), no bias, -> H ----
    for (int nt = 0; nt < 16; ++nt) {
      v8f acc = {};
      #pragma unroll
      for (int kc = 0; kc < 8; ++kc) {
        v16h a = loadA_lds(X, CAT_DIM, kc, lane);
        v16h b = loadB_glb(W + OFF_WCAT + (nt * 8 + kc) * 512, lane);
        acc = wmma16(a, b, acc);
      }
      int col = nt * 16 + m15;
      #pragma unroll
      for (int r = 0; r < 8; ++r)
        H[(r + 8 * g2) * INTER_DIM + col] = (_Float16)acc[r];
    }
    __syncthreads();

    // ---- mlp2 l1: H @ W1 + b -> X region (reused, stride 256) ----
    _Float16* Xr = X;
    for (int nt = 0; nt < 16; ++nt) {
      v8f acc = {};
      #pragma unroll
      for (int kc = 0; kc < 8; ++kc) {
        v16h a = loadA_lds(H, INTER_DIM, kc, lane);
        v16h b = loadB_glb(W + OFF_INT1 + (nt * 8 + kc) * 512, lane);
        acc = wmma16(a, b, acc);
      }
      int col = nt * 16 + m15;
      float bv = p.il1b[s][col];
      #pragma unroll
      for (int r = 0; r < 8; ++r)
        Xr[(r + 8 * g2) * INTER_DIM + col] = (_Float16)(acc[r] + bv);
    }
    __syncthreads();
    // LN + relu over 256
    if (lane < 16) {
      float s1 = 0.f, s2 = 0.f;
      for (int c = 0; c < 256; ++c) { float v = (float)Xr[lane * 256 + c]; s1 += v; s2 += v * v; }
      float mu = s1 * (1.f / 256.f), var = s2 * (1.f / 256.f) - mu * mu;
      float rs = rsqrtf(var + 1e-5f);
      for (int c = 0; c < 256; ++c) {
        float v = ((float)Xr[lane * 256 + c] - mu) * rs * p.ilng[s][c] + p.ilnb[s][c];
        Xr[lane * 256 + c] = (_Float16)(v > 0.f ? v : 0.f);
      }
    }
    __syncthreads();
    // ---- l2 (256->128) + bias, * sigmoid(gate), atomic scatter-add ----
    int dn[8];
    #pragma unroll
    for (int r = 0; r < 8; ++r) {
      int e = min(e0 + r + 8 * g2, p.E - 1);
      dn[r] = dIdx[e];
    }
    for (int nt = 0; nt < 8; ++nt) {
      v8f acc = {};
      #pragma unroll
      for (int kc = 0; kc < 8; ++kc) {
        v16h a = loadA_lds(Xr, INTER_DIM, kc, lane);
        v16h b = loadB_glb(W + OFF_INT2 + (nt * 8 + kc) * 512, lane);
        acc = wmma16(a, b, acc);
      }
      int col = nt * 16 + m15;
      float bv = p.il2b[s][col];
      #pragma unroll
      for (int r = 0; r < 8; ++r) {
        float msg = (acc[r] + bv) * gsig[nt][r];
        int row = r + 8 * g2;
        if (valid && (e0 + row) < p.E)
          atomicAdd(&accum[(size_t)dn[r] * 128 + col], msg);
      }
    }
    __syncthreads();  // protect X/H/G before next side restages
  }
}

struct P3 {
  const float *h_bond, *h_node;
  const int *left, *right;
  const _Float16 *w;
  const float *bsum[5];
  const float *o1b, *olng, *olnb, *o2b, *flng, *flnb;
  const float *accumL, *accumR;
  float *out;
  int E, numTiles;
};

__global__ void __launch_bounds__(32 * K3_WAVES) edge_update_kernel(P3 p) {
  __shared__ _Float16 sX[K3_WAVES][16 * 128];
  __shared__ _Float16 sH[K3_WAVES][16 * 128];
  __shared__ float    sF[K3_WAVES][16 * 128];
  __shared__ float    sMu[K3_WAVES][16];
  __shared__ float    sRs[K3_WAVES][16];
  const int lane = threadIdx.x & 31;
  const int wv = threadIdx.x >> 5;
  const int tile = blockIdx.x * K3_WAVES + wv;
  const bool valid = tile < p.numTiles;
  const int e0 = valid ? tile * 16 : 0;
  _Float16* X = sX[wv];
  _Float16* H = sH[wv];
  float* F = sF[wv];
  const int m15 = lane & 15, g2 = lane >> 4;

  v8f acc[8] = {};
  // five K=128 parts of the concatenated 640->128 matmul
  for (int pp = 0; pp < 5; ++pp) {
    for (int i = lane; i < 2048; i += 32) {
      int r = i >> 7, c = i & 127;
      int e = min(e0 + r, p.E - 1);
      float v;
      if (pp == 0)      v = p.accumL[(size_t)p.left[e]  * 128 + c];  // segsum(msgL,right)[left]
      else if (pp == 1) v = p.accumR[(size_t)p.right[e] * 128 + c];  // segsum(msgR,left)[right]
      else if (pp == 2) v = p.h_node[(size_t)p.left[e]  * 128 + c];
      else if (pp == 3) v = p.h_node[(size_t)p.right[e] * 128 + c];
      else { v = p.h_bond[(size_t)e * 128 + c]; F[i] = v; }          // also keep f32 residual
      X[i] = (_Float16)v;
    }
    __syncthreads();
    for (int nt = 0; nt < 8; ++nt) {
      #pragma unroll
      for (int kc = 0; kc < 4; ++kc) {
        v16h a = loadA_lds(X, 128, kc, lane);
        v16h b = loadB_glb(p.w + OFF_UPD + (nt * 20 + pp * 4 + kc) * 512, lane);
        acc[nt] = wmma16(a, b, acc[nt]);
      }
    }
    __syncthreads();
  }
  // + all five biases; store upd as f16 into X
  #pragma unroll
  for (int nt = 0; nt < 8; ++nt) {
    int col = nt * 16 + m15;
    float bv = p.bsum[0][col] + p.bsum[1][col] + p.bsum[2][col] + p.bsum[3][col] + p.bsum[4][col];
    #pragma unroll
    for (int r = 0; r < 8; ++r)
      X[(r + 8 * g2) * 128 + col] = (_Float16)(acc[nt][r] + bv);
  }
  __syncthreads();
  // out_layer l1
  for (int nt = 0; nt < 8; ++nt) {
    v8f h = {};
    #pragma unroll
    for (int kc = 0; kc < 4; ++kc) {
      v16h a = loadA_lds(X, 128, kc, lane);
      v16h b = loadB_glb(p.w + OFF_O1 + (nt * 4 + kc) * 512, lane);
      h = wmma16(a, b, h);
    }
    int col = nt * 16 + m15;
    float bv = p.o1b[col];
    #pragma unroll
    for (int r = 0; r < 8; ++r)
      H[(r + 8 * g2) * 128 + col] = (_Float16)(h[r] + bv);
  }
  __syncthreads();
  if (lane < 16) {
    float s1 = 0.f, s2 = 0.f;
    for (int c = 0; c < 128; ++c) { float v = (float)H[lane * 128 + c]; s1 += v; s2 += v * v; }
    float mu = s1 * (1.f / 128.f), var = s2 * (1.f / 128.f) - mu * mu;
    float rs = rsqrtf(var + 1e-5f);
    for (int c = 0; c < 128; ++c) {
      float v = ((float)H[lane * 128 + c] - mu) * rs * p.olng[c] + p.olnb[c];
      H[lane * 128 + c] = (_Float16)(v > 0.f ? v : 0.f);
    }
  }
  __syncthreads();
  // out_layer l2 + residual into F (f32)
  for (int nt = 0; nt < 8; ++nt) {
    v8f u = {};
    #pragma unroll
    for (int kc = 0; kc < 4; ++kc) {
      v16h a = loadA_lds(H, 128, kc, lane);
      v16h b = loadB_glb(p.w + OFF_O2 + (nt * 4 + kc) * 512, lane);
      u = wmma16(a, b, u);
    }
    int col = nt * 16 + m15;
    float bv = p.o2b[col];
    #pragma unroll
    for (int r = 0; r < 8; ++r) {
      int idx = (r + 8 * g2) * 128 + col;
      F[idx] = F[idx] + u[r] + bv;
    }
  }
  __syncthreads();
  if (lane < 16) {
    float s1 = 0.f, s2 = 0.f;
    for (int c = 0; c < 128; ++c) { float v = F[lane * 128 + c]; s1 += v; s2 += v * v; }
    float mu = s1 * (1.f / 128.f), var = s2 * (1.f / 128.f) - mu * mu;
    sMu[wv][lane] = mu;
    sRs[wv][lane] = rsqrtf(var + 1e-5f);
  }
  __syncthreads();
  for (int i = lane; i < 2048; i += 32) {
    int r = i >> 7, c = i & 127;
    int e = e0 + r;
    if (valid && e < p.E)
      p.out[(size_t)e * 128 + c] = (F[i] - sMu[wv][r]) * sRs[wv][r] * p.flng[c] + p.flnb[c];
  }
}

// Input flattening (setup_inputs insertion order, depth-first):
// 0 h_bond, 1 bond_index(int32 2xE), 2 h_node, 3 bond_extra,
// 4.. ffn_left {bond_w,node_w, inter{l1.w,l1.b,ln_g,ln_b,l2.w,l2.b}, gate{...}} (14 leaves)
// 18.. ffn_right (14), 32 msg_left.w, 33 .b, 34/35 msg_right, 36/37 node_ffn_left,
// 38/39 node_ffn_right, 40/41 self_ffn, 42 ln_g, 43 ln_b,
// 44..49 out_layer {l1.w,l1.b,ln_g,ln_b,l2.w,l2.b}
extern "C" void kernel_launch(void* const* d_in, const int* in_sizes, int n_in,
                              void* d_out, int out_size, void* d_ws, size_t ws_size,
                              hipStream_t stream) {
  const float* h_bond = (const float*)d_in[0];
  const int*   bidx   = (const int*)d_in[1];
  const float* h_node = (const float*)d_in[2];
  const float* extra  = (const float*)d_in[3];
  const int E  = in_sizes[0] / 128;
  const int NN = in_sizes[2] / 128;
  const int* left  = bidx;
  const int* right = bidx + E;

  float* accumL = (float*)d_ws;
  float* accumR = accumL + (size_t)NN * 128;
  _Float16* wpk = (_Float16*)(accumR + (size_t)NN * 128);

  {
    size_t zn = (size_t)NN * 256;
    zero_f32<<<(unsigned)((zn + 255) / 256), 256, 0, stream>>>(accumL, zn);
  }

  auto PK = [&](int i, int Ks, int Nn, int off, int ktc, int kbo) {
    int tot = Ks * Nn;
    pack_w<<<(tot + 255) / 256, 256, 0, stream>>>((const float*)d_in[i],
                                                  wpk + off, Ks, Nn, ktc, kbo);
  };
  for (int s = 0; s < 2; ++s) {
    int b = 4 + s * 14, off = s * SIDE_STRIDE;
    PK(b + 0, 128, 256, off + OFF_WCAT, 8, 0);   // bond_w  -> Wcat rows 0..127
    PK(b + 1, 128, 256, off + OFF_WCAT, 8, 4);   // node_w  -> Wcat rows 128..255
    PK(b + 2, 256, 256, off + OFF_INT1, 8, 0);   // inter.l1.w
    PK(b + 6, 256, 128, off + OFF_INT2, 8, 0);   // inter.l2.w
    PK(b + 8, 288, 32,  off + OFF_G1,   9, 0);   // gate.l1.w
    PK(b + 12, 32, 128, off + OFF_G2,   1, 0);   // gate.l2.w
  }
  PK(32, 128, 128, OFF_UPD, 20, 0);   // msg_left.w
  PK(34, 128, 128, OFF_UPD, 20, 4);   // msg_right.w
  PK(36, 128, 128, OFF_UPD, 20, 8);   // node_ffn_left.w
  PK(38, 128, 128, OFF_UPD, 20, 12);  // node_ffn_right.w
  PK(40, 128, 128, OFF_UPD, 20, 16);  // self_ffn.w
  PK(44, 128, 128, OFF_O1, 4, 0);     // out_layer.l1.w
  PK(48, 128, 128, OFF_O2, 4, 0);     // out_layer.l2.w

  const int tiles = (E + 15) / 16;

  P2 p2;
  p2.h_bond = h_bond; p2.h_node = h_node; p2.extra = extra;
  p2.left = left; p2.right = right; p2.w = wpk;
  for (int s = 0; s < 2; ++s) {
    int b = 4 + s * 14;
    p2.il1b[s] = (const float*)d_in[b + 3];
    p2.ilng[s] = (const float*)d_in[b + 4];
    p2.ilnb[s] = (const float*)d_in[b + 5];
    p2.il2b[s] = (const float*)d_in[b + 7];
    p2.gl1b[s] = (const float*)d_in[b + 9];
    p2.glng[s] = (const float*)d_in[b + 10];
    p2.glnb[s] = (const float*)d_in[b + 11];
    p2.gl2b[s] = (const float*)d_in[b + 13];
  }
  p2.accumL = accumL; p2.accumR = accumR; p2.E = E; p2.numTiles = tiles;
  edge_msg_kernel<<<(tiles + K2_WAVES - 1) / K2_WAVES, 32 * K2_WAVES, 0, stream>>>(p2);

  P3 p3;
  p3.h_bond = h_bond; p3.h_node = h_node; p3.left = left; p3.right = right; p3.w = wpk;
  p3.bsum[0] = (const float*)d_in[33];
  p3.bsum[1] = (const float*)d_in[35];
  p3.bsum[2] = (const float*)d_in[37];
  p3.bsum[3] = (const float*)d_in[39];
  p3.bsum[4] = (const float*)d_in[41];
  p3.o1b  = (const float*)d_in[45];
  p3.olng = (const float*)d_in[46];
  p3.olnb = (const float*)d_in[47];
  p3.o2b  = (const float*)d_in[49];
  p3.flng = (const float*)d_in[42];
  p3.flnb = (const float*)d_in[43];
  p3.accumL = accumL; p3.accumR = accumR;
  p3.out = (float*)d_out; p3.E = E; p3.numTiles = tiles;
  edge_update_kernel<<<(tiles + K3_WAVES - 1) / K3_WAVES, 32 * K3_WAVES, 0, stream>>>(p3);

  (void)n_in; (void)out_size; (void)ws_size;
}
